// GAT_60163901882507
// MI455X (gfx1250) — matrix operand order repaired and checked by reading, not compiled, not run on hardware
//
#include <hip/hip_runtime.h>

typedef __attribute__((ext_vector_type(16))) _Float16 v16h;
typedef __attribute__((ext_vector_type(8)))  float    v8f;

#define NEG 0.2f
#define EPSV 1e-16f

__device__ __forceinline__ float lrelu(float v) { return v > 0.f ? v : NEG * v; }

// float atomic max via signed/unsigned int atomics (works for mixed signs, init -inf)
__device__ __forceinline__ void atomicMaxF(float* a, float v) {
  if (v >= 0.f) atomicMax((int*)a, __float_as_int(v));
  else          atomicMin((unsigned int*)a, (unsigned int)__float_as_int(v));
}

// ---------------- conversions / fills ----------------
__global__ void f32_to_f16_kernel(const float* __restrict__ in, _Float16* __restrict__ out, int n) {
  int i = blockIdx.x * blockDim.x + threadIdx.x;
  if (i < n) out[i] = (_Float16)in[i];
}

__global__ void fill_f32_kernel(float* __restrict__ p, float v, int n) {
  int i = blockIdx.x * blockDim.x + threadIdx.x;
  if (i < n) p[i] = v;
}

// Pack B[K,Nc] (f32 row-major) into WMMA B-fragment-major f16:
//   P[((ktile*(Nc/16) + ntile)*32 + lane)*16 + j] = B[(ktile*32 + kb + (j&7) + (j>>3)*16)*Nc + ntile*16 + n]
// with n = lane&15, kb = (lane>>4)*8  (matches the 32x16 f16 B layout).
// Each lane's 16 halves become 32 contiguous bytes -> two b128 loads in the GEMM.
__global__ void pack_b_kernel(const float* __restrict__ B, _Float16* __restrict__ P,
                              int K, int Nc) {
  int idx = blockIdx.x * blockDim.x + threadIdx.x;
  if (idx >= K * Nc) return;
  int j    = idx & 15;
  int lane = (idx >> 4) & 31;
  int rest = idx >> 9;
  int ntiles = Nc >> 4;
  int nt = rest % ntiles;
  int kt = rest / ntiles;
  int n  = lane & 15;
  int kb = (lane >> 4) << 3;
  P[idx] = (_Float16)B[(size_t)(kt * 32 + kb + (j & 7) + ((j >> 3) << 4)) * Nc + nt * 16 + n];
}

// ---------------- WMMA GEMM: C[M,Nc] = A[M,K] @ Bpacked, f16 in, f32 out ----------------
// A-frag (16x32 f16 ISA layout): lane r=lane&15 (row), hi=lane>>4;
//   halves 0..7  -> K = (hi?8:0)      + 0..7
//   halves 8..15 -> K = (hi?8:0) + 16 + 0..7    (two contiguous 16B loads)
__device__ __forceinline__ v16h load_a_frag(const _Float16* A, int lda) {
  int lane = threadIdx.x & 31;
  const _Float16* p = A + (lane & 15) * lda + (lane >> 4) * 8;
  union { v16h v; float4 f[2]; } u;
  u.f[0] = *(const float4*)(p);
  u.f[1] = *(const float4*)(p + 16);
  return u.v;
}

__device__ __forceinline__ v16h load_b_frag_packed(const _Float16* P) {
  int lane = threadIdx.x & 31;
  const _Float16* p = P + lane * 16;   // 32B contiguous per lane, 1KB per wave
  union { v16h v; float4 f[2]; } u;
  u.f[0] = *(const float4*)(p);
  u.f[1] = *(const float4*)(p + 8);
  return u.v;
}

template <int NT, int KK, int NC>
__global__ __launch_bounds__(256) void gemm_wmma_f16(
    const _Float16* __restrict__ A, const _Float16* __restrict__ Bp,
    float* __restrict__ C, int M) {
  constexpr int NTILES  = NC / 16;
  constexpr int NGROUPS = NTILES / NT;
  int wave = (blockIdx.x * blockDim.x + threadIdx.x) >> 5;
  int lane = threadIdx.x & 31;
  int mtiles = M >> 4;                  // M multiple of 16 (50000 = 3125*16)
  if (wave >= mtiles * NGROUPS) return; // wave-uniform exit (EXEC all-ones inside)
  int tm  = wave % mtiles;
  int tn0 = (wave / mtiles) * NT;
  const _Float16* Abase = A + (size_t)tm * 16 * KK;
  v8f acc[NT];
#pragma unroll
  for (int i = 0; i < NT; ++i)
#pragma unroll
    for (int k = 0; k < 8; ++k) acc[i][k] = 0.f;
#pragma unroll
  for (int kt = 0; kt < KK / 32; ++kt) {
    v16h a = load_a_frag(Abase + kt * 32, KK);
#pragma unroll
    for (int i = 0; i < NT; ++i) {
      v16h b = load_b_frag_packed(Bp + ((size_t)kt * NTILES + (tn0 + i)) * 512);
      acc[i] = __builtin_amdgcn_wmma_f32_16x16x32_f16(
          false, a, false, b, (short)0, acc[i], false, false);
    }
  }
  // C/D layout: lanes 0-15: N=lane, VGPR v -> M=v ; lanes 16-31: N=lane-16, M=v+8
  int n = lane & 15;
  int hi = lane >> 4;
#pragma unroll
  for (int i = 0; i < NT; ++i) {
    float* Cb = C + (size_t)tm * 16 * NC + (size_t)(tn0 + i) * 16 + n;
#pragma unroll
    for (int v = 0; v < 8; ++v) Cb[(size_t)(v + hi * 8) * NC] = acc[i][v];
  }
}

// ---------------- conv1 edge kernels (H=8, C=32, row width 256) ----------------
// wave per edge; lane covers 8 consecutive channels -> head = lane>>2
__global__ void edge1_logits(const float* __restrict__ XL, const float* __restrict__ XR,
                             const float* __restrict__ att, const int* __restrict__ ei,
                             float* __restrict__ alpha, float* __restrict__ amax,
                             int E, int N) {
  int gid = blockIdx.x * blockDim.x + threadIdx.x;
  int wave = gid >> 5, lane = gid & 31;
  int Etot = E + N;
  if (wave >= Etot) return;
  int s = wave < E ? ei[wave]     : wave - E;
  int d = wave < E ? ei[E + wave] : wave - E;
  const float4* pl = (const float4*)(XL + (size_t)s * 256) + lane * 2;
  const float4* pr = (const float4*)(XR + (size_t)d * 256) + lane * 2;
  const float4* pa = (const float4*)(att) + lane * 2;
  float part = 0.f;
#pragma unroll
  for (int t = 0; t < 2; ++t) {
    float4 l = pl[t], r = pr[t], a = pa[t];
    part += a.x * lrelu(l.x + r.x) + a.y * lrelu(l.y + r.y) +
            a.z * lrelu(l.z + r.z) + a.w * lrelu(l.w + r.w);
  }
  part += __shfl_xor(part, 1, 32);
  part += __shfl_xor(part, 2, 32);
  if ((lane & 3) == 0) {
    int h = lane >> 2;
    alpha[(size_t)wave * 8 + h] = part;
    atomicMaxF(amax + d * 8 + h, part);
  }
}

__global__ void edge1_exp(const int* __restrict__ ei, float* __restrict__ alpha,
                          const float* __restrict__ amax, float* __restrict__ denom,
                          int E, int N) {
  int e = blockIdx.x * blockDim.x + threadIdx.x;
  int Etot = E + N;
  if (e >= Etot) return;
  int d = e < E ? ei[E + e] : e - E;
#pragma unroll
  for (int h = 0; h < 8; ++h) {
    float v = __expf(alpha[(size_t)e * 8 + h] - amax[d * 8 + h]);
    alpha[(size_t)e * 8 + h] = v;
    atomicAdd(denom + d * 8 + h, v);
  }
}

__global__ void edge1_aggregate(const float* __restrict__ XL, const float* __restrict__ alpha,
                                const int* __restrict__ ei, float* __restrict__ agg,
                                int E, int N) {
  int gid = blockIdx.x * blockDim.x + threadIdx.x;
  int wave = gid >> 5, lane = gid & 31;
  int Etot = E + N;
  if (wave >= Etot) return;
  int s = wave < E ? ei[wave]     : wave - E;
  int d = wave < E ? ei[E + wave] : wave - E;
  const float* pl = XL + (size_t)s * 256;
  float* po = agg + (size_t)d * 256;
  float w0 = alpha[(size_t)wave * 8 + (lane >> 3)];       // channels [lane*4, ..): head lane/8
  float w1 = alpha[(size_t)wave * 8 + 4 + (lane >> 3)];   // channels 128+lane*4: head 4+lane/8
  float4 x0 = *(const float4*)(pl + lane * 4);
  float4 x1 = *(const float4*)(pl + 128 + lane * 4);
  atomicAdd(po + lane * 4 + 0, x0.x * w0);
  atomicAdd(po + lane * 4 + 1, x0.y * w0);
  atomicAdd(po + lane * 4 + 2, x0.z * w0);
  atomicAdd(po + lane * 4 + 3, x0.w * w0);
  atomicAdd(po + 128 + lane * 4 + 0, x1.x * w1);
  atomicAdd(po + 128 + lane * 4 + 1, x1.y * w1);
  atomicAdd(po + 128 + lane * 4 + 2, x1.z * w1);
  atomicAdd(po + 128 + lane * 4 + 3, x1.w * w1);
}

// h1 = elu(agg/(denom+eps) + b1), emitted as f16 for the next WMMA GEMM
__global__ void node1_finalize(const float* __restrict__ agg, const float* __restrict__ den,
                               const float* __restrict__ b1, _Float16* __restrict__ h16, int N) {
  int i = blockIdx.x * blockDim.x + threadIdx.x;
  if (i >= N * 256) return;
  int n = i >> 8, j = i & 255;
  float v = agg[i] / (den[n * 8 + (j >> 5)] + EPSV) + b1[j];
  v = v > 0.f ? v : __expf(v) - 1.f;
  h16[i] = (_Float16)v;
}

// ---------------- conv2 edge kernels (H=1, C=32) ----------------
__global__ void edge2_logits(const float* __restrict__ XL, const float* __restrict__ XR,
                             const float* __restrict__ att, const int* __restrict__ ei,
                             float* __restrict__ alpha, float* __restrict__ amax,
                             int E, int N) {
  int gid = blockIdx.x * blockDim.x + threadIdx.x;
  int wave = gid >> 5, lane = gid & 31;
  int Etot = E + N;
  if (wave >= Etot) return;
  int s = wave < E ? ei[wave]     : wave - E;
  int d = wave < E ? ei[E + wave] : wave - E;
  float v = XL[(size_t)s * 32 + lane] + XR[(size_t)d * 32 + lane];
  float part = lrelu(v) * att[lane];
  part += __shfl_xor(part, 16, 32);
  part += __shfl_xor(part, 8, 32);
  part += __shfl_xor(part, 4, 32);
  part += __shfl_xor(part, 2, 32);
  part += __shfl_xor(part, 1, 32);
  if (lane == 0) {
    alpha[wave] = part;
    atomicMaxF(amax + d, part);
  }
}

__global__ void edge2_exp(const int* __restrict__ ei, float* __restrict__ alpha,
                          const float* __restrict__ amax, float* __restrict__ denom,
                          int E, int N) {
  int e = blockIdx.x * blockDim.x + threadIdx.x;
  int Etot = E + N;
  if (e >= Etot) return;
  int d = e < E ? ei[E + e] : e - E;
  float v = __expf(alpha[e] - amax[d]);
  alpha[e] = v;
  atomicAdd(denom + d, v);
}

__global__ void edge2_aggregate(const float* __restrict__ XL, const float* __restrict__ alpha,
                                const int* __restrict__ ei, float* __restrict__ agg,
                                int E, int N) {
  int gid = blockIdx.x * blockDim.x + threadIdx.x;
  int wave = gid >> 5, lane = gid & 31;
  int Etot = E + N;
  if (wave >= Etot) return;
  int s = wave < E ? ei[wave]     : wave - E;
  int d = wave < E ? ei[E + wave] : wave - E;
  atomicAdd(agg + (size_t)d * 32 + lane, XL[(size_t)s * 32 + lane] * alpha[wave]);
}

__global__ void node2_finalize(const float* __restrict__ agg, const float* __restrict__ den,
                               const float* __restrict__ b2, float* __restrict__ h2, int N) {
  int i = blockIdx.x * blockDim.x + threadIdx.x;
  if (i >= N * 32) return;
  h2[i] = agg[i] / (den[i >> 5] + EPSV) + b2[i & 31];
}

// ---------------- output: logits = h2@Wout + bout, log_softmax ----------------
__global__ void out_logsoftmax(const float* __restrict__ h2, const float* __restrict__ Wout,
                               const float* __restrict__ bout, float* __restrict__ out, int N) {
  int n = blockIdx.x * blockDim.x + threadIdx.x;
  if (n >= N) return;
  float h[32];
  const float4* ph = (const float4*)(h2 + (size_t)n * 32);
#pragma unroll
  for (int t = 0; t < 8; ++t) {
    float4 v = ph[t];
    h[t * 4] = v.x; h[t * 4 + 1] = v.y; h[t * 4 + 2] = v.z; h[t * 4 + 3] = v.w;
  }
  float* po = out + (size_t)n * 40;
  float m = -__builtin_inff();
  for (int j = 0; j < 40; ++j) {
    float acc = bout[j];
#pragma unroll
    for (int c = 0; c < 32; ++c) acc += h[c] * Wout[c * 40 + j];
    po[j] = acc;
    m = fmaxf(m, acc);
  }
  float ssum = 0.f;
  for (int j = 0; j < 40; ++j) ssum += __expf(po[j] - m);
  float lse = m + logf(ssum);
  for (int j = 0; j < 40; ++j) po[j] -= lse;
}

// ---------------- host ----------------
static inline unsigned nblk(size_t n) { return (unsigned)((n + 255) / 256); }

extern "C" void kernel_launch(void* const* d_in, const int* in_sizes, int n_in,
                              void* d_out, int out_size, void* d_ws, size_t ws_size,
                              hipStream_t stream) {
  const float* x    = (const float*)d_in[0];
  const int*   ei   = (const int*)d_in[1];
  const float* W1l  = (const float*)d_in[2];
  const float* W1r  = (const float*)d_in[3];
  const float* att1 = (const float*)d_in[4];
  const float* b1   = (const float*)d_in[5];
  const float* W2l  = (const float*)d_in[6];
  const float* W2r  = (const float*)d_in[7];
  const float* att2 = (const float*)d_in[8];
  const float* b2   = (const float*)d_in[9];
  const float* Wout = (const float*)d_in[10];
  const float* bout = (const float*)d_in[11];
  float* out = (float*)d_out;

  const int N = in_sizes[0] / 128;  // 50000
  const int E = in_sizes[1] / 2;    // 800000
  const int Etot = E + N;

  // workspace carve-out (phase-2 buffers alias dead phase-1 buffers)
  char* w = (char*)d_ws;
  size_t off = 0;
  auto take = [&](size_t bytes) -> void* {
    void* p = w + off;
    off += (bytes + 255) & ~(size_t)255;
    return p;
  };
  _Float16* xf16  = (_Float16*)take((size_t)N * 128 * 2);
  _Float16* w1l16 = (_Float16*)take(128 * 256 * 2);   // packed B fragments
  _Float16* w1r16 = (_Float16*)take(128 * 256 * 2);
  _Float16* w2l16 = (_Float16*)take(256 * 32 * 2);
  _Float16* w2r16 = (_Float16*)take(256 * 32 * 2);
  float* XL1    = (float*)take((size_t)N * 256 * 4);
  float* XR1    = (float*)take((size_t)N * 256 * 4);
  float* AGG1   = (float*)take((size_t)N * 256 * 4);
  float* alpha1 = (float*)take((size_t)Etot * 8 * 4);
  float* amax1  = (float*)take((size_t)N * 8 * 4);
  float* den1   = (float*)take((size_t)N * 8 * 4);
  _Float16* h1f16 = (_Float16*)take((size_t)N * 256 * 2);
  // phase-2 aliases (lifetimes disjoint from original uses)
  float* XL2    = (float*)xf16;                               // 6.4MB of 12.8MB
  float* XR2    = (float*)((char*)xf16 + (size_t)N * 32 * 4); // second half
  float* alpha2 = XL1;
  float* amax2  = amax1;
  float* den2   = den1;
  float* AGG2   = XR1;
  float* h2     = AGG1;

  const dim3 B256(256);
  const float ninf = -__builtin_inff();

  // --- convert activations to f16; pack weights into B-fragment layout ---
  f32_to_f16_kernel<<<nblk((size_t)N * 128), B256, 0, stream>>>(x, xf16, N * 128);
  pack_b_kernel<<<nblk(128 * 256), B256, 0, stream>>>(W1l, w1l16, 128, 256);
  pack_b_kernel<<<nblk(128 * 256), B256, 0, stream>>>(W1r, w1r16, 128, 256);
  pack_b_kernel<<<nblk(256 * 32), B256, 0, stream>>>(W2l, w2l16, 256, 32);
  pack_b_kernel<<<nblk(256 * 32), B256, 0, stream>>>(W2r, w2r16, 256, 32);

  // --- conv1 GEMMs: XL1/XR1 = x @ W1l/W1r  (M=50000, K=128, Nc=256) ---
  {
    size_t waves = (size_t)(N / 16) * (256 / 64);
    gemm_wmma_f16<4, 128, 256><<<nblk(waves * 32), B256, 0, stream>>>(xf16, w1l16, XL1, N);
    gemm_wmma_f16<4, 128, 256><<<nblk(waves * 32), B256, 0, stream>>>(xf16, w1r16, XR1, N);
  }

  // --- conv1 edge softmax + aggregation ---
  fill_f32_kernel<<<nblk((size_t)N * 8), B256, 0, stream>>>(amax1, ninf, N * 8);
  fill_f32_kernel<<<nblk((size_t)N * 8), B256, 0, stream>>>(den1, 0.f, N * 8);
  fill_f32_kernel<<<nblk((size_t)N * 256), B256, 0, stream>>>(AGG1, 0.f, N * 256);
  edge1_logits<<<nblk((size_t)Etot * 32), B256, 0, stream>>>(XL1, XR1, att1, ei, alpha1, amax1, E, N);
  edge1_exp<<<nblk((size_t)Etot), B256, 0, stream>>>(ei, alpha1, amax1, den1, E, N);
  edge1_aggregate<<<nblk((size_t)Etot * 32), B256, 0, stream>>>(XL1, alpha1, ei, AGG1, E, N);
  node1_finalize<<<nblk((size_t)N * 256), B256, 0, stream>>>(AGG1, den1, b1, h1f16, N);

  // --- conv2 GEMMs: XL2/XR2 = h1 @ W2l/W2r  (M=50000, K=256, Nc=32) ---
  {
    size_t waves = (size_t)(N / 16);
    gemm_wmma_f16<2, 256, 32><<<nblk(waves * 32), B256, 0, stream>>>(h1f16, w2l16, XL2, N);
    gemm_wmma_f16<2, 256, 32><<<nblk(waves * 32), B256, 0, stream>>>(h1f16, w2r16, XR2, N);
  }

  // --- conv2 edge softmax + aggregation ---
  fill_f32_kernel<<<nblk((size_t)N), B256, 0, stream>>>(amax2, ninf, N);
  fill_f32_kernel<<<nblk((size_t)N), B256, 0, stream>>>(den2, 0.f, N);
  fill_f32_kernel<<<nblk((size_t)N * 32), B256, 0, stream>>>(AGG2, 0.f, N * 32);
  edge2_logits<<<nblk((size_t)Etot * 32), B256, 0, stream>>>(XL2, XR2, att2, ei, alpha2, amax2, E, N);
  edge2_exp<<<nblk((size_t)Etot), B256, 0, stream>>>(ei, alpha2, amax2, den2, E, N);
  edge2_aggregate<<<nblk((size_t)Etot * 32), B256, 0, stream>>>(XL2, alpha2, ei, AGG2, E, N);
  node2_finalize<<<nblk((size_t)N * 32), B256, 0, stream>>>(AGG2, den2, b2, h2, N);

  // --- output projection + log_softmax ---
  out_logsoftmax<<<nblk((size_t)N), B256, 0, stream>>>(h2, Wout, bout, out, N);
}
